// WindowedMultiHeadAttention_43310450213358
// MI455X (gfx1250) — matrix operand order repaired and checked by reading, not compiled, not run on hardware
//
#include <hip/hip_runtime.h>

// ---------------------------------------------------------------------------
// Windowed MHA (Swin) fused kernel for gfx1250 (MI455X).
// One workgroup (256 threads = 8 wave32) per 8x8 window. All intermediates in
// LDS; matrix math via v_wmma_f32_16x16x32_f16 with f32 accumulation.
// ---------------------------------------------------------------------------

typedef __attribute__((ext_vector_type(16))) _Float16 v16h;
typedef __attribute__((ext_vector_type(8)))  float    v8f;

#define WS_   8
#define LQ    64          // tokens per window
#define CIN   384
#define CQKV  1152
#define NHEAD 12
#define DH    32
#define HIMG  128
#define WIMG  128

// ---- LDS layout (units: _Float16 elements) --------------------------------
#define QS_STRIDE 1168                       // 1152 + pad (16B-multiple rows)
#define QS_OFF    0
#define QS_HALFS  (LQ * QS_STRIDE)           // 74752
#define XS_STRIDE 392                        // 384 + pad
#define XS_OFF    QS_HALFS                   // overlaps S-scratch (Xs dies first)
#define SS_OFF    QS_HALFS
#define SS_STRIDE 72                         // 64 + pad
#define SS_WAVE   (LQ * SS_STRIDE)           // 4608 per wave
#define OS_OFF    (QS_HALFS + 8 * SS_WAVE)   // 111616
#define OS_STRIDE 392
#define LDS_HALFS (OS_OFF + LQ * OS_STRIDE)  // 136704 halfs = 273408 B < 320 KB

// ---- WMMA fragment loaders (contiguous-K order, wave32) -------------------
// A (16x32 f16): lanes 0-15 -> M=0..15, K = 0..15 ; lanes 16-31 same M,
// K = 16..31.
__device__ __forceinline__ v16h load_a_frag(const _Float16* base, int stride,
                                            int row0, int k0, int lane) {
  const int m     = lane & 15;
  const int kbase = (lane >> 4) * 16;
  const _Float16* p = base + (row0 + m) * stride + k0 + kbase;
  v16h a;
#pragma unroll
  for (int e = 0; e < 8; ++e) a[e] = p[e];
#pragma unroll
  for (int e = 0; e < 8; ++e) a[8 + e] = p[8 + e];
  return a;
}

// B (32x16 f16), source rows hold the K-dimension contiguously:
// B[k][n] = src[(row0 + n)*stride + c0 + k]; lanes 0-15 -> K=0..15, 16-31 -> K=16..31.
__device__ __forceinline__ v16h load_b_frag_contig(const _Float16* base, int stride,
                                                   int row0, int c0, int lane) {
  const int n      = lane & 15;
  const int kstart = (lane >> 4) * 16;
  const _Float16* p = base + (row0 + n) * stride + c0 + kstart;
  v16h b;
#pragma unroll
  for (int e = 0; e < 16; ++e) b[e] = p[e];
  return b;
}

// B (32x16 f16) where K runs down rows: B[k][n] = src[(k0+k)*stride + col0 + n].
__device__ __forceinline__ v16h load_b_frag_strided(const _Float16* base, int stride,
                                                    int col0, int k0, int lane) {
  const int n      = (lane & 15);
  const int kstart = (lane >> 4) * 16;
  v16h b;
#pragma unroll
  for (int e = 0; e < 16; ++e) b[e] = base[(k0 + kstart + e) * stride + col0 + n];
  return b;
}

__device__ __forceinline__ v8f wmma16x16x32(v16h a, v16h b, v8f c) {
  return __builtin_amdgcn_wmma_f32_16x16x32_f16(false, a, false, b,
                                                (short)0, c, false, false);
}

// ---------------------------------------------------------------------------
__global__ void cvt_f32_f16_kernel(const float* __restrict__ src,
                                   _Float16* __restrict__ dst, int n) {
  int i = blockIdx.x * blockDim.x + threadIdx.x;
  if (i < n) dst[i] = (_Float16)src[i];
}

// ---------------------------------------------------------------------------
__global__ void __launch_bounds__(256)
swin_window_attn_kernel(const float* __restrict__ x,
                        const _Float16* __restrict__ wqkv,   // [1152][384] f16
                        const float* __restrict__ bqkv,      // [1152]
                        const _Float16* __restrict__ wproj,  // [384][384] f16
                        const float* __restrict__ bproj,     // [384]
                        const float* __restrict__ bias_table,// [225][12]
                        const int* __restrict__ rel_idx,     // [64*64]
                        float* __restrict__ out) {
  extern __shared__ _Float16 lds[];
  _Float16* Qs = lds + QS_OFF;  // qkv  [64][1168]
  _Float16* Xs = lds + XS_OFF;  // x    [64][392]   (stages 1-2)
  _Float16* Ss = lds + SS_OFF;  // S    per-wave [64][72] (stage 3)
  _Float16* Os = lds + OS_OFF;  // attn out [64][392]

  const int tid  = threadIdx.x;
  const int lane = tid & 31;
  const int wave = tid >> 5;

  const int blk = blockIdx.x;         // 2048 windows
  const int b   = blk >> 8;           // batch
  const int rem = blk & 255;
  const int wh  = rem >> 4;           // window row  (16)
  const int ww  = rem & 15;           // window col  (16)

  // ---- Stage 1: window partition, f32 -> f16 into LDS ---------------------
  for (int i = tid; i < LQ * CIN; i += 256) {
    int c = i >> 6;                   // channel
    int l = i & 63;                   // token
    int r = l >> 3, col = l & 7;
    float v = x[((b * CIN + c) * HIMG + wh * WS_ + r) * WIMG + ww * WS_ + col];
    Xs[l * XS_STRIDE + c] = (_Float16)v;
  }
  __syncthreads();

  // ---- Stage 2: qkv = Xs @ Wqkv^T + bqkv  (4 x 72 tiles of 16x16) ---------
  for (int t = wave; t < 4 * (CQKV / 16); t += 8) {
    const int mt = t / (CQKV / 16);
    const int nt = t % (CQKV / 16);
    const int o  = nt * 16 + (lane & 15);
    v8f acc = {};
    for (int ks = 0; ks < CIN; ks += 32) {
      __builtin_prefetch(wqkv + (nt * 16 + (lane & 15)) * CIN + ks + 64, 0, 0);
      v16h a  = load_a_frag(Xs, XS_STRIDE, mt * 16, ks, lane);
      v16h bf = load_b_frag_contig(wqkv, CIN, nt * 16, ks, lane);
      acc = wmma16x16x32(a, bf, acc);
    }
    const float bias = bqkv[o];
    const int mrow = mt * 16 + (lane >> 4) * 8;
#pragma unroll
    for (int rr = 0; rr < 8; ++rr)
      Qs[(mrow + rr) * QS_STRIDE + o] = (_Float16)(acc[rr] + bias);
  }
  __syncthreads();

  // ---- Stage 3: attention per head (head index wave-uniform) --------------
  const float scale = 0.17677669529663687f;  // 1/sqrt(32)
  for (int h = wave; h < NHEAD; h += 8) {
    _Float16* Sw = Ss + wave * SS_WAVE;
    const int qcol = h * DH;
    const int kcol = CIN + h * DH;
    const int vcol = 2 * CIN + h * DH;

    // S = (q @ k^T) * scale + rel_pos_bias   (4x4 tiles, K = 32 = one WMMA)
    for (int mt = 0; mt < 4; ++mt)
      for (int nt = 0; nt < 4; ++nt) {
        v8f acc = {};
        v16h a  = load_a_frag(Qs, QS_STRIDE, mt * 16, qcol, lane);
        v16h bf = load_b_frag_contig(Qs, QS_STRIDE, nt * 16, kcol, lane);
        acc = wmma16x16x32(a, bf, acc);
        const int mrow = mt * 16 + (lane >> 4) * 8;
        const int n    = nt * 16 + (lane & 15);
#pragma unroll
        for (int rr = 0; rr < 8; ++rr) {
          const int l  = mrow + rr;
          const int ri = rel_idx[l * LQ + n];
          const float bv = bias_table[ri * NHEAD + h];
          Sw[l * SS_STRIDE + n] = (_Float16)(acc[rr] * scale + bv);
        }
      }

    // Row softmax (2 rows per lane, f32 math, exp recomputed to save VGPRs)
#pragma unroll
    for (int rr = 0; rr < 2; ++rr) {
      const int l = lane + rr * 32;
      _Float16* row = Sw + l * SS_STRIDE;
      float mx = -1e30f;
      for (int m = 0; m < LQ; ++m) { float v = (float)row[m]; mx = v > mx ? v : mx; }
      float sum = 0.f;
      for (int m = 0; m < LQ; ++m) sum += __expf((float)row[m] - mx);
      const float inv = 1.f / sum;
      for (int m = 0; m < LQ; ++m)
        row[m] = (_Float16)(__expf((float)row[m] - mx) * inv);
    }

    // O_head = P @ V   (4x2 tiles, K = 64 in two WMMA steps)
    for (int mt = 0; mt < 4; ++mt)
      for (int nt = 0; nt < 2; ++nt) {
        v8f acc = {};
        for (int ks = 0; ks < LQ; ks += 32) {
          v16h a  = load_a_frag(Sw, SS_STRIDE, mt * 16, ks, lane);
          v16h bf = load_b_frag_strided(Qs, QS_STRIDE, vcol + nt * 16, ks, lane);
          acc = wmma16x16x32(a, bf, acc);
        }
        const int mrow = mt * 16 + (lane >> 4) * 8;
        const int n    = nt * 16 + (lane & 15);
#pragma unroll
        for (int rr = 0; rr < 8; ++rr)
          Os[(mrow + rr) * OS_STRIDE + h * DH + n] = (_Float16)acc[rr];
      }
  }
  __syncthreads();

  // ---- Stage 4: out = Os @ Wproj^T + bproj -> global NCHW -----------------
  for (int t = wave; t < 4 * (CIN / 16); t += 8) {
    const int mt = t / (CIN / 16);
    const int nt = t % (CIN / 16);
    const int o  = nt * 16 + (lane & 15);
    v8f acc = {};
    for (int ks = 0; ks < CIN; ks += 32) {
      __builtin_prefetch(wproj + (nt * 16 + (lane & 15)) * CIN + ks + 64, 0, 0);
      v16h a  = load_a_frag(Os, OS_STRIDE, mt * 16, ks, lane);
      v16h bf = load_b_frag_contig(wproj, CIN, nt * 16, ks, lane);
      acc = wmma16x16x32(a, bf, acc);
    }
    const float bias = bproj[o];
    const int mrow = mt * 16 + (lane >> 4) * 8;
#pragma unroll
    for (int rr = 0; rr < 8; ++rr) {
      const int l = mrow + rr;
      const int r = l >> 3, col = l & 7;
      out[((b * CIN + o) * HIMG + wh * WS_ + r) * WIMG + ww * WS_ + col] =
          acc[rr] + bias;
    }
  }
}

// ---------------------------------------------------------------------------
extern "C" void kernel_launch(void* const* d_in, const int* in_sizes, int n_in,
                              void* d_out, int out_size, void* d_ws, size_t ws_size,
                              hipStream_t stream) {
  (void)in_sizes; (void)n_in; (void)out_size; (void)ws_size;

  const float* x        = (const float*)d_in[0];
  const float* w_qkv    = (const float*)d_in[1];
  const float* b_qkv    = (const float*)d_in[2];
  const float* w_proj   = (const float*)d_in[3];
  const float* b_proj   = (const float*)d_in[4];
  const float* bias_tab = (const float*)d_in[5];
  const int*   rel_idx  = (const int*)d_in[6];
  float* out = (float*)d_out;

  // Workspace: f16 copies of the weight matrices (1.2 MB, L2-resident).
  _Float16* wqkv_h  = (_Float16*)d_ws;
  _Float16* wproj_h = wqkv_h + (size_t)CQKV * CIN;

  const int n1 = CQKV * CIN;   // 442368
  cvt_f32_f16_kernel<<<(n1 + 255) / 256, 256, 0, stream>>>(w_qkv, wqkv_h, n1);
  const int n2 = CIN * CIN;    // 147456
  cvt_f32_f16_kernel<<<(n2 + 255) / 256, 256, 0, stream>>>(w_proj, wproj_h, n2);

  const size_t lds_bytes = (size_t)LDS_HALFS * sizeof(_Float16);  // 273408 B
  swin_window_attn_kernel<<<dim3(2048), dim3(256), lds_bytes, stream>>>(
      x, wqkv_h, b_qkv, wproj_h, b_proj, bias_tab, rel_idx, out);
}
